// SlotESGSingleTaskSimple_49254684951085
// MI455X (gfx1250) — compile-verified, matrix operand order checked by reading, not run on hardware
//
#include <hip/hip_runtime.h>
#include <math.h>

typedef __attribute__((ext_vector_type(16))) _Float16 v16h;
typedef __attribute__((ext_vector_type(8)))  float    v8f;

// ---------------- problem constants ----------------
#define BB   512
#define KT   64
#define MM   4
#define DD   512
#define LL   8
#define DPR  64
#define DFI  128
#define DEV  32
#define DNW  768
#define XNW  992   // 64+128+32+768

// ---------------- f16 workspace layout (element offsets) ----------------
#define WENC_OFF  0ull                                   // 512*992 = 507904
#define QW_OFF    507904ull
#define KW_OFF    (QW_OFF + 262144ull)
#define VW_OFF    (KW_OFF + 262144ull)
#define WIH_OFF   (VW_OFF + 262144ull)                   // 1536*512
#define WHH_OFF   (WIH_OFF + 786432ull)
#define F1W_OFF   (WHH_OFF + 786432ull)
#define F2W_OFF   (F1W_OFF + 262144ull)
#define FUSW_OFF  (F2W_OFF + 262144ull)                  // 512*4096
#define XN_OFF    (FUSW_OFF + 2097152ull)                // 32768*992
#define H_OFF     (XN_OFF + 32505856ull)                 // 32768*4*512
#define SL_OFF    (H_OFF + 67108864ull)                  // 512*4096
#define WS_F16_END (SL_OFF + 2097152ull)
#define FV_BYTE_OFF (WS_F16_END * 2ull)                  // float Fv[512*512]

// ---- slot kernel dynamic-LDS pool layout (bytes) ----
#define AB_BYTE    16384u     // Aa [16][512] f16 at 0, Ab [16][512] f16 at 16384
#define CREG_BYTE  32768u     // SNEW f32 [16][512] (32KB)  ALIASES  Qh(16K)+Kh(8K)+Vh(8K)
#define ATTN_BYTE  65536u     // attnS float[64]
#define POOL_BYTES 65792u

// ---------------- helpers ----------------
__device__ __forceinline__ float wred(float v) {
  #pragma unroll
  for (int o = 16; o > 0; o >>= 1) v += __shfl_xor(v, o, 32);
  return v;
}

__device__ __forceinline__ float fixval(float v) {
  if (isnan(v)) return 0.f;
  if (isinf(v)) return v > 0.f ? 1e5f : -1e5f;
  return v;
}

__device__ __forceinline__ float sigm(float x) { return 1.f / (1.f + expf(-x)); }

__device__ __forceinline__ unsigned pcg(unsigned v) {
  unsigned s = v * 747796405u + 2891336453u;
  unsigned w = ((s >> ((s >> 28) + 4u)) ^ s) * 277803737u;
  return (w >> 22) ^ w;
}

// 16-bit 16x32 A-fragment (or mirrored 32x16 B-fragment) from row-major memory:
// lane<16 -> row=lane, K 0..7 & 16..23 ; lane>=16 -> row=lane-16, K 8..15 & 24..31.
__device__ __forceinline__ v16h load_frag16(const _Float16* base, int stride) {
  int lane = threadIdx.x & 31;
  int r  = lane & 15;
  int ko = (lane >> 4) << 3;
  const _Float16* p = base + r * stride + ko;
  v16h f;
  #pragma unroll
  for (int i = 0; i < 8; ++i) { f[i] = p[i]; f[i + 8] = p[i + 16]; }
  return f;
}

__device__ __forceinline__ v8f wmma32(v16h a, v16h b, v8f c) {
  return __builtin_amdgcn_wmma_f32_16x16x32_f16(false, a, false, b, (short)0, c, false, false);
}

// CDNA5 async global->LDS copy (ASYNCcnt path, CDNA5 ISA)
__device__ __forceinline__ void async_load_b128(unsigned lds_byte_addr, const void* gaddr) {
  asm volatile("global_load_async_to_lds_b128 %0, %1, off"
               :: "v"(lds_byte_addr), "v"(gaddr) : "memory");
}
__device__ __forceinline__ void wait_async0() {
#if __has_builtin(__builtin_amdgcn_s_wait_asynccnt)
  __builtin_amdgcn_s_wait_asynccnt(0);
#else
  asm volatile("s_wait_asynccnt 0" ::: "memory");
#endif
}

// ---------------- kernel 1: weight f32 -> f16 ----------------
__global__ void cvt_f32_to_f16(const float* __restrict__ src, _Float16* __restrict__ dst, int n) {
  int i = blockIdx.x * blockDim.x + threadIdx.x;
  int stride = gridDim.x * blockDim.x;
  for (; i < n; i += stride) dst[i] = (_Float16)src[i];
}

// ---------------- kernel 2: per-modality LayerNorm -> f16 Xn ----------------
__global__ __launch_bounds__(256)
void ln_kernel(const float* __restrict__ price, const float* __restrict__ fin,
               const float* __restrict__ ev,    const float* __restrict__ nw,
               const float* pw, const float* pb, const float* fw, const float* fb,
               const float* ew, const float* eb, const float* nww, const float* nwb,
               _Float16* __restrict__ Xn) {
  int wave = blockIdx.x * 8 + (threadIdx.x >> 5);
  int lane = threadIdx.x & 31;
  int bk = wave >> 2, m = wave & 3;
  if (bk >= BB * KT) return;
  const float *x, *w, *bi; int d, xoff;
  if (m == 0)      { x = price + (size_t)bk * DPR; w = pw;  bi = pb;  d = DPR; xoff = 0;   }
  else if (m == 1) { x = fin   + (size_t)bk * DFI; w = fw;  bi = fb;  d = DFI; xoff = 64;  }
  else if (m == 2) { x = ev    + (size_t)bk * DEV; w = ew;  bi = eb;  d = DEV; xoff = 192; }
  else             { x = nw    + (size_t)bk * DNW; w = nww; bi = nwb; d = DNW; xoff = 224; }
  float s1 = 0.f, s2 = 0.f;
  for (int i = lane; i < d; i += 32) {
    float v = fixval(x[i]); s1 += v; s2 += v * v;
  }
  s1 = wred(s1); s2 = wred(s2);
  float mean = s1 / d;
  float var  = fmaxf(s2 / d - mean * mean, 0.f);
  float rsq  = rsqrtf(var + 1e-5f);
  _Float16* out = Xn + (size_t)bk * XNW + xoff;
  for (int i = lane; i < d; i += 32) {
    float v = fixval(x[i]);
    float y = (v - mean) * rsq * w[i] + bi[i];
    y = fminf(fmaxf(y, -1000.f), 1000.f);
    out[i] = (_Float16)y;
  }
}

// ---------------- kernel 3: encoder GEMM (WMMA) -> f16 H ----------------
__global__ __launch_bounds__(256)
void enc_gemm_kernel(const _Float16* __restrict__ Xn, const _Float16* __restrict__ Wenc,
                     const float* bp, const float* bf, const float* be, const float* bn,
                     _Float16* __restrict__ H) {
  int m = blockIdx.y;
  int bk0 = blockIdx.x * 16;
  int lane = threadIdx.x & 31, wv = threadIdx.x >> 5;
  int dm, xoff; size_t woff; const float* bias;
  if (m == 0)      { dm = DPR; xoff = 0;   woff = 0;       bias = bp; }
  else if (m == 1) { dm = DFI; xoff = 64;  woff = 32768;   bias = bf; }
  else if (m == 2) { dm = DEV; xoff = 192; woff = 98304;   bias = be; }
  else             { dm = DNW; xoff = 224; woff = 114688;  bias = bn; }
  const _Float16* Abase = Xn + (size_t)bk0 * XNW + xoff;
  const _Float16* Wm    = Wenc + woff;
  #pragma unroll
  for (int j = 0; j < 4; ++j) {
    int n0 = (wv * 4 + j) * 16;
    v8f acc = {};
    for (int k0 = 0; k0 < dm; k0 += 32) {
      v16h a   = load_frag16(Abase + k0, XNW);
      v16h bfr = load_frag16(Wm + (size_t)n0 * dm + k0, dm);
      acc = wmma32(a, bfr, acc);
    }
    int n = n0 + (lane & 15);
    int rbase = (lane >> 4) * 8;
    float bv = bias[n];
    #pragma unroll
    for (int r = 0; r < 8; ++r) {
      int row = bk0 + rbase + r;
      float v = fmaxf(acc[r] + bv, 0.f);
      H[((size_t)row * MM + m) * DD + n] = (_Float16)v;
    }
  }
}

// ---------------- kernel 4: recurrent slot attention (2 batch elems / block) ----------------
__global__ __launch_bounds__(256)
void slot_kernel(const _Float16* __restrict__ ws16,
                 const float* qb, const float* kb, const float* vb,
                 const float* bih, const float* bhh,
                 const float* f1b, const float* f2b,
                 const float* lnw, const float* lnb,
                 const float* mu, const float* lsig,
                 float* __restrict__ S_out, float* __restrict__ Beta_out) {
  const _Float16* QW  = ws16 + QW_OFF;
  const _Float16* KW  = ws16 + KW_OFF;
  const _Float16* VW  = ws16 + VW_OFF;
  const _Float16* WIH = ws16 + WIH_OFF;
  const _Float16* WHH = ws16 + WHH_OFF;
  const _Float16* F1W = ws16 + F1W_OFF;
  const _Float16* F2W = ws16 + F2W_OFF;
  const _Float16* H16 = ws16 + H_OFF;

  int b0 = blockIdx.x * 2;
  int tid = threadIdx.x, lane = tid & 31, wv = tid >> 5;

  extern __shared__ __attribute__((aligned(16))) char pool[];
  _Float16* Aa    = (_Float16*)pool;                       // [16][512]: slots b0 rows 0-7, b1 rows 8-15; later h1
  _Float16* Ab    = (_Float16*)(pool + AB_BYTE);           // [16][512]: h_t (rows 0-7 + zero) / upd (16) / sNew f16
  float*    SNEW  = (float*)(pool + CREG_BYTE);            // [16][512] f32  (aliases Qh/Kh/Vh)
  _Float16* Qh    = (_Float16*)(pool + CREG_BYTE);         // [16][512]
  _Float16* Kh    = (_Float16*)(pool + CREG_BYTE + 16384); // [8][512]  row = bb*4+m
  _Float16* Vh    = (_Float16*)(pool + CREG_BYTE + 24576); // [8][512]
  float*    attnS = (float*)(pool + ATTN_BYTE);            // [16][4]

  // init slots0 = mu + exp(log_sigma)*eps for both batch elems
  for (int i = tid; i < 2 * LL * DD; i += 256) {
    int bb = i >> 12, j = i & 4095;
    unsigned base = (unsigned)((b0 + bb) * 4096 + j);
    float u1 = (pcg(base * 2u + 1u) + 1u) * 2.3283064e-10f;
    float u2 = pcg(base * 2u + 2u) * 2.3283064e-10f;
    float eps = sqrtf(-2.f * logf(u1)) * cosf(6.2831853f * u2);
    Aa[i] = (_Float16)(mu[j] + expf(lsig[j]) * eps);
  }
  __syncthreads();

  const float scale = 0.044194173824159216f; // 1/sqrt(512)

  for (int t = 0; t < KT; ++t) {
    // ---- phase 0: async h_t -> Ab rows 0-7 (b0 rows 0-3, b1 rows 4-7); zero rows 8-15
    {
      const _Float16* h0 = H16 + ((size_t)((b0 + 0) * KT + t) * MM) * DD;
      const _Float16* h1 = H16 + ((size_t)((b0 + 1) * KT + t) * MM) * DD;
      unsigned abbase = (unsigned)(uintptr_t)(void*)Ab;
      async_load_b128(abbase + (unsigned)tid * 16u,          h0 + tid * 8);
      async_load_b128(abbase + 4096u + (unsigned)tid * 16u,  h1 + tid * 8);
      for (int i = tid; i < 2048; i += 256) ((unsigned*)(Ab + 4096))[i] = 0u;
      if (t + 1 < KT && tid < 64) {  // warm next step's h_t
        const _Float16* hn = H16 + ((size_t)((b0 + ((tid >> 5) & 1)) * KT + (t + 1)) * MM) * DD;
        __builtin_prefetch(hn + (tid & 31) * 64, 0, 1);
      }
      wait_async0();
    }
    __syncthreads();

    // ---- Q = slots @ qW^T + qb  (16 rows, all valid)
    #pragma unroll
    for (int j = 0; j < 4; ++j) {
      int n0 = (wv * 4 + j) * 16;
      v8f acc = {};
      for (int k0 = 0; k0 < DD; k0 += 32) {
        v16h a = load_frag16(Aa + k0, DD);
        v16h w = load_frag16(QW + (size_t)n0 * DD + k0, DD);
        acc = wmma32(a, w, acc);
      }
      int n = n0 + (lane & 15);
      int rbase = (lane >> 4) * 8;
      float bv = qb[n];
      #pragma unroll
      for (int r = 0; r < 8; ++r) Qh[(rbase + r) * DD + n] = (_Float16)(acc[r] + bv);
    }
    // ---- K,V = h_t @ {kW,vW}^T + b   (rows 0-7 valid: bb*4+m)
    #pragma unroll
    for (int j = 0; j < 4; ++j) {
      int n0 = (wv * 4 + j) * 16;
      v8f aK = {}, aV = {};
      for (int k0 = 0; k0 < DD; k0 += 32) {
        v16h a   = load_frag16(Ab + k0, DD);
        v16h wk  = load_frag16(KW + (size_t)n0 * DD + k0, DD);
        v16h wvv = load_frag16(VW + (size_t)n0 * DD + k0, DD);
        aK = wmma32(a, wk, aK);
        aV = wmma32(a, wvv, aV);
      }
      if (lane < 16) {
        int n = n0 + lane; float bk1 = kb[n], bv1 = vb[n];
        #pragma unroll
        for (int r = 0; r < 8; ++r) {
          Kh[r * DD + n] = (_Float16)(aK[r] + bk1);
          Vh[r * DD + n] = (_Float16)(aV[r] + bv1);
        }
      }
    }
    __syncthreads();

    // ---- scores (64 pairs) + softmax
    #pragma unroll
    for (int jj = 0; jj < 8; ++jj) {
      int p = wv * 8 + jj;        // p = (bb*8+l)*4 + m
      int gl = p >> 2, m = p & 3;
      int kr = ((gl >> 3) << 2) + m;
      float s = 0.f;
      for (int i = lane; i < DD; i += 32)
        s += (float)Qh[gl * DD + i] * (float)Kh[kr * DD + i];
      s = wred(s);
      if (lane == 0) attnS[p] = s * scale;
    }
    __syncthreads();
    if (tid < 16) {
      int gl = tid;
      float v0 = attnS[gl*4+0], v1 = attnS[gl*4+1], v2 = attnS[gl*4+2], v3 = attnS[gl*4+3];
      float mx = fmaxf(fmaxf(v0, v1), fmaxf(v2, v3));
      float e0 = expf(v0 - mx), e1 = expf(v1 - mx), e2 = expf(v2 - mx), e3 = expf(v3 - mx);
      float sum = e0 + e1 + e2 + e3;
      float a0 = e0 / sum, a1 = e1 / sum, a2 = e2 / sum, a3 = e3 / sum;
      if (isnan(a0)) a0 = 0.f; if (isnan(a1)) a1 = 0.f;
      if (isnan(a2)) a2 = 0.f; if (isnan(a3)) a3 = 0.f;
      attnS[gl*4+0] = a0; attnS[gl*4+1] = a1; attnS[gl*4+2] = a2; attnS[gl*4+3] = a3;
      float* bo = Beta_out + (((size_t)(b0 + (gl >> 3)) * KT + t) * LL + (gl & 7)) * MM;
      bo[0] = a0; bo[1] = a1; bo[2] = a2; bo[3] = a3;
    }
    __syncthreads();

    // ---- upd = attn @ V -> Ab rows 0-15 (f16)
    for (int i = tid; i < 2 * LL * DD; i += 256) {
      int gr = i >> 9, n = i & 511, bb = gr >> 3;
      const _Float16* vrow = Vh + (bb << 2) * DD + n;
      float u = attnS[gr*4+0] * (float)vrow[0]
              + attnS[gr*4+1] * (float)vrow[DD]
              + attnS[gr*4+2] * (float)vrow[2*DD]
              + attnS[gr*4+3] * (float)vrow[3*DD];
      Ab[i] = (_Float16)u;
    }
    __syncthreads();

    // ---- GRU: 6-chain fused GEMM -> SNEW (16 rows)
    #pragma unroll
    for (int j = 0; j < 4; ++j) {
      int n0 = (wv * 4 + j) * 16;
      v8f aI0 = {}, aI1 = {}, aI2 = {}, aH0 = {}, aH1 = {}, aH2 = {};
      for (int k0 = 0; k0 < DD; k0 += 32) {
        v16h aU = load_frag16(Ab + k0, DD);
        v16h aS = load_frag16(Aa + k0, DD);
        aI0 = wmma32(aU, load_frag16(WIH + ((size_t)(0*DD + n0)) * DD + k0, DD), aI0);
        aI1 = wmma32(aU, load_frag16(WIH + ((size_t)(1*DD + n0)) * DD + k0, DD), aI1);
        aI2 = wmma32(aU, load_frag16(WIH + ((size_t)(2*DD + n0)) * DD + k0, DD), aI2);
        aH0 = wmma32(aS, load_frag16(WHH + ((size_t)(0*DD + n0)) * DD + k0, DD), aH0);
        aH1 = wmma32(aS, load_frag16(WHH + ((size_t)(1*DD + n0)) * DD + k0, DD), aH1);
        aH2 = wmma32(aS, load_frag16(WHH + ((size_t)(2*DD + n0)) * DD + k0, DD), aH2);
      }
      int n = n0 + (lane & 15);
      int rbase = (lane >> 4) * 8;
      float bi0 = bih[n], bi1 = bih[DD + n], bi2 = bih[2*DD + n];
      float bh0 = bhh[n], bh1 = bhh[DD + n], bh2 = bhh[2*DD + n];
      #pragma unroll
      for (int r = 0; r < 8; ++r) {
        int gr = rbase + r;
        float rg = sigm((aI0[r] + bi0) + (aH0[r] + bh0));
        float z  = sigm((aI1[r] + bi1) + (aH1[r] + bh1));
        float nn = tanhf((aI2[r] + bi2) + rg * (aH2[r] + bh2));
        float sold = (float)Aa[gr * DD + n];
        SNEW[gr * DD + n] = (1.f - z) * nn + z * sold;
      }
    }
    __syncthreads();

    // ---- FFN layer 1: h1 = relu(sNew @ f1W^T + f1b) -> Aa (16 rows)
    for (int i = tid; i < 2 * LL * DD; i += 256) Ab[i] = (_Float16)SNEW[i];
    __syncthreads();
    #pragma unroll
    for (int j = 0; j < 4; ++j) {
      int n0 = (wv * 4 + j) * 16;
      v8f acc = {};
      for (int k0 = 0; k0 < DD; k0 += 32)
        acc = wmma32(load_frag16(Ab + k0, DD),
                     load_frag16(F1W + (size_t)n0 * DD + k0, DD), acc);
      int n = n0 + (lane & 15);
      int rbase = (lane >> 4) * 8;
      float bv = f1b[n];
      #pragma unroll
      for (int r = 0; r < 8; ++r)
        Aa[(rbase + r) * DD + n] = (_Float16)fmaxf(acc[r] + bv, 0.f);
    }
    __syncthreads();
    // ---- FFN layer 2 + residual: SNEW += h1 @ f2W^T + f2b
    #pragma unroll
    for (int j = 0; j < 4; ++j) {
      int n0 = (wv * 4 + j) * 16;
      v8f acc = {};
      for (int k0 = 0; k0 < DD; k0 += 32)
        acc = wmma32(load_frag16(Aa + k0, DD),
                     load_frag16(F2W + (size_t)n0 * DD + k0, DD), acc);
      int n = n0 + (lane & 15);
      int rbase = (lane >> 4) * 8;
      float bv = f2b[n];
      #pragma unroll
      for (int r = 0; r < 8; ++r)
        SNEW[(rbase + r) * DD + n] += acc[r] + bv;
    }
    __syncthreads();

    // ---- slot LayerNorm (16 rows), write S, refresh Aa f16 slots
    #pragma unroll
    for (int rr = 0; rr < 2; ++rr) {
      int gr = wv + rr * 8;
      float s1 = 0.f, s2 = 0.f;
      for (int i = lane; i < DD; i += 32) {
        float v = SNEW[gr * DD + i]; s1 += v; s2 += v * v;
      }
      s1 = wred(s1); s2 = wred(s2);
      float mean = s1 / DD;
      float var  = fmaxf(s2 / DD - mean * mean, 0.f);
      float rsq  = rsqrtf(var + 1e-5f);
      float* so = S_out + (((size_t)(b0 + (gr >> 3)) * KT + t) * LL + (gr & 7)) * DD;
      for (int i = lane; i < DD; i += 32) {
        float y = (SNEW[gr * DD + i] - mean) * rsq * lnw[i] + lnb[i];
        so[i] = y;
        Aa[gr * DD + i] = (_Float16)y;
      }
    }
    __syncthreads();
  }
}

// ---------------- kernel 5: S[:, -1] f32 -> f16 ----------------
__global__ void slast_cvt(const float* __restrict__ S, _Float16* __restrict__ SL, int n) {
  int i = blockIdx.x * blockDim.x + threadIdx.x;
  int stride = gridDim.x * blockDim.x;
  for (; i < n; i += stride) {
    int b = i >> 12, j = i & 4095;
    SL[i] = (_Float16)S[(((size_t)b * KT + (KT - 1)) * LL) * DD + j];
  }
}

// ---------------- kernel 6: fusion GEMM ----------------
__global__ __launch_bounds__(256)
void fus_gemm_kernel(const _Float16* __restrict__ SL, const _Float16* __restrict__ FUSW,
                     const float* fusb, float* __restrict__ Fv) {
  int row0 = blockIdx.x * 16;
  int lane = threadIdx.x & 31, wv = threadIdx.x >> 5;
  #pragma unroll
  for (int j = 0; j < 4; ++j) {
    int n0 = (wv * 4 + j) * 16;
    v8f acc = {};
    for (int k0 = 0; k0 < 4096; k0 += 32) {
      v16h a = load_frag16(SL + (size_t)row0 * 4096 + k0, 4096);
      v16h w = load_frag16(FUSW + (size_t)n0 * 4096 + k0, 4096);
      acc = wmma32(a, w, acc);
    }
    int n = n0 + (lane & 15);
    int rbase = (lane >> 4) * 8;
    float bv = fusb[n];
    #pragma unroll
    for (int r = 0; r < 8; ++r)
      Fv[(size_t)(row0 + rbase + r) * DD + n] = fmaxf(acc[r] + bv, 0.f);
  }
}

// ---------------- kernel 7: head ----------------
__global__ __launch_bounds__(256)
void head_kernel(const float* __restrict__ Fv, const float* hw, const float* hb,
                 float* __restrict__ pred) {
  int wave = blockIdx.x * 8 + (threadIdx.x >> 5);
  int lane = threadIdx.x & 31;
  if (wave >= BB) return;
  float s = 0.f;
  for (int i = lane; i < DD; i += 32) s += Fv[(size_t)wave * DD + i] * hw[i];
  s = wred(s);
  if (lane == 0) pred[wave] = 1.f / (1.f + expf(-(s + hb[0])));
}

// ---------------- launcher ----------------
extern "C" void kernel_launch(void* const* d_in, const int* in_sizes, int n_in,
                              void* d_out, int out_size, void* d_ws, size_t ws_size,
                              hipStream_t stream) {
  (void)in_sizes; (void)n_in; (void)out_size; (void)ws_size;
  const float* price = (const float*)d_in[0];
  const float* fin   = (const float*)d_in[1];
  const float* ev    = (const float*)d_in[2];
  const float* nw    = (const float*)d_in[3];
  const float* price_ln_w = (const float*)d_in[4];
  const float* price_ln_b = (const float*)d_in[5];
  const float* price_W    = (const float*)d_in[6];
  const float* price_b    = (const float*)d_in[7];
  const float* fin_ln_w   = (const float*)d_in[8];
  const float* fin_ln_b   = (const float*)d_in[9];
  const float* fin_W      = (const float*)d_in[10];
  const float* fin_b      = (const float*)d_in[11];
  const float* ev_ln_w    = (const float*)d_in[12];
  const float* ev_ln_b    = (const float*)d_in[13];
  const float* ev_W       = (const float*)d_in[14];
  const float* ev_b       = (const float*)d_in[15];
  const float* nw_ln_w    = (const float*)d_in[16];
  const float* nw_ln_b    = (const float*)d_in[17];
  const float* nw_W       = (const float*)d_in[18];
  const float* nw_b       = (const float*)d_in[19];
  const float* mu         = (const float*)d_in[20];
  const float* log_sigma  = (const float*)d_in[21];
  const float* q_W = (const float*)d_in[22]; const float* q_b = (const float*)d_in[23];
  const float* k_W = (const float*)d_in[24]; const float* k_b = (const float*)d_in[25];
  const float* v_W = (const float*)d_in[26]; const float* v_b = (const float*)d_in[27];
  const float* W_ih = (const float*)d_in[28]; const float* W_hh = (const float*)d_in[29];
  const float* b_ih = (const float*)d_in[30]; const float* b_hh = (const float*)d_in[31];
  const float* f1_W = (const float*)d_in[32]; const float* f1_b = (const float*)d_in[33];
  const float* f2_W = (const float*)d_in[34]; const float* f2_b = (const float*)d_in[35];
  const float* sln_w = (const float*)d_in[36]; const float* sln_b = (const float*)d_in[37];
  const float* fus_W = (const float*)d_in[38]; const float* fus_b = (const float*)d_in[39];
  const float* head_W = (const float*)d_in[40]; const float* head_b = (const float*)d_in[41];

  _Float16* ws16 = (_Float16*)d_ws;
  float* Fv = (float*)((char*)d_ws + FV_BYTE_OFF);

  float* pred = (float*)d_out;                              // [512]
  float* S    = pred + 512;                                 // [512,64,8,512]
  float* Beta = S + (size_t)BB * KT * LL * DD;              // [512,64,8,4]

  auto cvt = [&](const float* src, size_t dstOff, int n) {
    int grid = (n + 255) / 256;
    cvt_f32_to_f16<<<grid, 256, 0, stream>>>(src, ws16 + dstOff, n);
  };
  cvt(price_W, WENC_OFF + 0,      512 * DPR);
  cvt(fin_W,   WENC_OFF + 32768,  512 * DFI);
  cvt(ev_W,    WENC_OFF + 98304,  512 * DEV);
  cvt(nw_W,    WENC_OFF + 114688, 512 * DNW);
  cvt(q_W,  QW_OFF,  512 * 512);
  cvt(k_W,  KW_OFF,  512 * 512);
  cvt(v_W,  VW_OFF,  512 * 512);
  cvt(W_ih, WIH_OFF, 1536 * 512);
  cvt(W_hh, WHH_OFF, 1536 * 512);
  cvt(f1_W, F1W_OFF, 512 * 512);
  cvt(f2_W, F2W_OFF, 512 * 512);
  cvt(fus_W, FUSW_OFF, 512 * 4096);

  ln_kernel<<<(BB * KT * MM) / 8, 256, 0, stream>>>(
      price, fin, ev, nw,
      price_ln_w, price_ln_b, fin_ln_w, fin_ln_b,
      ev_ln_w, ev_ln_b, nw_ln_w, nw_ln_b,
      ws16 + XN_OFF);

  enc_gemm_kernel<<<dim3((BB * KT) / 16, MM), 256, 0, stream>>>(
      ws16 + XN_OFF, ws16 + WENC_OFF, price_b, fin_b, ev_b, nw_b, ws16 + H_OFF);

  slot_kernel<<<BB / 2, 256, POOL_BYTES, stream>>>(
      ws16, q_b, k_b, v_b, b_ih, b_hh, f1_b, f2_b, sln_w, sln_b,
      mu, log_sigma, S, Beta);

  slast_cvt<<<(BB * LL * DD + 255) / 256, 256, 0, stream>>>(S, ws16 + SL_OFF, BB * LL * DD);
  fus_gemm_kernel<<<BB / 16, 256, 0, stream>>>(ws16 + SL_OFF, ws16 + FUSW_OFF, fus_b, Fv);
  head_kernel<<<BB / 8, 256, 0, stream>>>(Fv, head_W, head_b, pred);
}
